// SatelliteImageGNN_5506148073796
// MI455X (gfx1250) — matrix operand order repaired and checked by reading, not compiled, not run on hardware
//
#include <hip/hip_runtime.h>
#include <hip/hip_bf16.h>

// Problem constants (match the reference).
#define HH 768           // low-res grid side
#define SS 3             // pixel-shuffle scale
#define HID 32           // hidden channels
#define TILE 16          // output tile side per workgroup
#define HALO 18          // TILE + 2 (1-pixel halo for the 3x3 stencil)
#define NHALO 324        // HALO*HALO
#define ROWS_PAD 336     // ceil(NHALO/16)*16  -> WMMA row tiles
#define MT 21            // ROWS_PAD/16
#define NODE_PAD 337     // odd padding for bank-conflict-free LDS columns

typedef __attribute__((ext_vector_type(16))) _Float16 v16h;
typedef __attribute__((ext_vector_type(8)))  float    v8f;
typedef int vint4 __attribute__((vector_size(16)));   // matches builtin param

#if defined(__has_builtin)
#if __has_builtin(__builtin_amdgcn_global_load_async_to_lds_b128)
#define HAVE_ASYNC_LDS 1
#endif
#if __has_builtin(__builtin_amdgcn_s_wait_asynccnt)
#define HAVE_WAIT_ASYNC 1
#endif
#endif

// dinv[i] = rsqrt(deg_i + 1); deg+1 == clamped 3x3 window area.
__device__ __forceinline__ float dinv_at(int gi, int gj) {
    int rs = 1 + (gi > 0) + (gi < HH - 1);
    int cs = 1 + (gj > 0) + (gj < HH - 1);
    return rsqrtf((float)(rs * cs));
}

// One fused GCN layer over a 16x16 node tile.
// Intermediates in global ws are stored PRE-SCALED by dinv[node] (f16), so
// f16-input layers do a raw 16B copy into LDS (async-to-LDS when available).
template <int CIN, int COUT, bool IN_F16, bool RELU, bool PIXSHUF>
__global__ __launch_bounds__(256) void gcn_layer_kernel(
    const void* __restrict__ in, const float* __restrict__ W,
    const float* __restrict__ bias, void* __restrict__ out)
{
    __shared__ _Float16 haloH[ROWS_PAD][HID];   // 21.5 KB, K-padded to 32
    __shared__ float    tbuf[HID][NODE_PAD];    // 43.1 KB, channel-major

    const int tid = threadIdx.x;
    const int bi = blockIdx.y, bj = blockIdx.x;

    // ---- Phase 1: halo -> LDS ----
    if (IN_F16) {
        // Pure copy: 4 x 16B chunks per halo row; values already dinv-scaled.
        for (int u = tid; u < ROWS_PAD * 4; u += 256) {
            int hidx = u >> 2, chunk = u & 3;
            int hr = hidx / HALO, hc = hidx % HALO;
            int gi = bi * TILE + hr - 1;
            int gj = bj * TILE + hc - 1;
            bool valid = (hidx < NHALO) && gi >= 0 && gi < HH && gj >= 0 && gj < HH;
            uint4* ldst = (uint4*)&haloH[hidx][chunk * 8];
            if (valid) {
                const uint4* gsrc =
                    (const uint4*)in + ((size_t)(gi * HH + gj) * 4 + chunk);
#ifdef HAVE_ASYNC_LDS
                __builtin_amdgcn_global_load_async_to_lds_b128(
                    (__attribute__((address_space(1))) vint4*)(void*)
                        const_cast<uint4*>(gsrc),
                    (__attribute__((address_space(3))) vint4*)(void*)ldst,
                    /*offset=*/0, /*cpol=*/0);
#else
                *ldst = *gsrc;
#endif
            } else {
                uint4 z = {0u, 0u, 0u, 0u};
                *ldst = z;
            }
        }
#ifdef HAVE_ASYNC_LDS
#ifdef HAVE_WAIT_ASYNC
        __builtin_amdgcn_s_wait_asynccnt(0);
#else
        asm volatile("s_wait_asynccnt 0x0" ::: "memory");
#endif
#endif
    } else {
        // f32 input (layer 1): scale by dinv[j], convert to f16, K-pad to 32.
        for (int hidx = tid; hidx < ROWS_PAD; hidx += 256) {
            int hr = hidx / HALO, hc = hidx % HALO;
            int gi = bi * TILE + hr - 1;
            int gj = bj * TILE + hc - 1;
            bool valid = (hidx < NHALO) && gi >= 0 && gi < HH && gj >= 0 && gj < HH;
            float dj = 0.0f;
            long  base = 0;
            if (valid) {
                dj = dinv_at(gi, gj);
                base = (long)(gi * HH + gj) * CIN;
            }
            #pragma unroll
            for (int c = 0; c < HID; ++c) {
                float v = 0.0f;
                if (valid && c < CIN) v = ((const float*)in)[base + c] * dj;
                haloH[hidx][c] = (_Float16)v;
            }
        }
    }
    __syncthreads();

    // ---- Phase 2: t = haloH @ W via v_wmma_f32_16x16x32_f16 (one K step) ----
    const int lane = tid & 31;                                   // wave32
    const int wave = __builtin_amdgcn_readfirstlane(tid) >> 5;   // scalar wave id
    constexpr int NT = (COUT + 15) / 16;                         // column tiles
    const int ntile = (NT == 2) ? (wave & 1) : 0;                // scalar

    // B fragment: 32x16 f16; lane%16 = column, lanes 0-15 K=0-7,16-23, 16-31 K=8-15,24-31
    v16h bfrag;
    {
        int n  = ntile * 16 + (lane & 15);
        int kb = (lane < 16) ? 0 : 8;
        #pragma unroll
        for (int e = 0; e < 16; ++e) {
            int k = kb + ((e < 8) ? e : 8 + e);
            float w = (k < CIN && n < COUT) ? W[k * COUT + n] : 0.0f;
            bfrag[e] = (_Float16)w;
        }
    }

    const int mstart = (NT == 2) ? (wave >> 1) : wave;   // scalar
    const int mstep  = (NT == 2) ? 4 : 8;                // scalar
    for (int mt = mstart; mt < MT; mt += mstep) {
        // A per ISA 16-bit 16x32 layout: lanes 0-15 row M=lane (K 0-7,16-23),
        // lanes 16-31 row M=lane-16 (K 8-15,24-31) -> two contiguous 32B halves.
        int row  = mt * 16 + (lane & 15);
        int koff = (lane < 16) ? 0 : 8;
        v16h afrag;
        #pragma unroll
        for (int v = 0; v < 8; ++v) {
            int kp = ((v < 4) ? 2 * v : 16 + 2 * (v - 4)) + koff;
            afrag[2 * v]     = haloH[row][kp];
            afrag[2 * v + 1] = haloH[row][kp + 1];
        }
        v8f acc = {};
        acc = __builtin_amdgcn_wmma_f32_16x16x32_f16(
            /*neg_a=*/false, afrag, /*neg_b=*/false, bfrag,
            /*c_mod=*/(short)0, acc, /*reuse_a=*/false, /*reuse_b=*/false);

        // D layout: VGPR r -> (M = r + 8*(lane>=16), N = lane%16)
        int nodebase = mt * 16 + ((lane < 16) ? 0 : 8);
        int ch = ntile * 16 + (lane & 15);
        #pragma unroll
        for (int r = 0; r < 8; ++r)
            tbuf[ch][nodebase + r] = acc[r];
    }
    __syncthreads();

    // ---- Phase 3: 3x3 box sum, * dinv[i], + bias, relu, store ----
    const int tx = tid & 15, ty = tid >> 4;
    const int gi = bi * TILE + ty, gj = bj * TILE + tx;
    const float di = dinv_at(gi, gj);
    const int hb = (ty + 1) * HALO + (tx + 1);
    #pragma unroll
    for (int ch = 0; ch < COUT; ++ch) {
        float s = 0.0f;
        #pragma unroll
        for (int dr = -1; dr <= 1; ++dr)
            #pragma unroll
            for (int dc = -1; dc <= 1; ++dc)
                s += tbuf[ch][hb + dr * HALO + dc];
        float val = s * di + bias[ch];
        if (RELU) val = fmaxf(val, 0.0f);
        if (PIXSHUF) {
            int si = ch / SS, sj = ch % SS;   // [N, S*S] -> (1, H*S, H*S)
            ((float*)out)[(long)(gi * SS + si) * (HH * SS) + (gj * SS + sj)] = val;
        } else {
            // Store pre-scaled by dinv[i] so the next layer does a raw copy.
            ((_Float16*)out)[(long)(gi * HH + gj) * HID + ch] =
                (_Float16)(val * di);
        }
    }
}

extern "C" void kernel_launch(void* const* d_in, const int* in_sizes, int n_in,
                              void* d_out, int out_size, void* d_ws, size_t ws_size,
                              hipStream_t stream) {
    (void)in_sizes; (void)n_in; (void)out_size; (void)ws_size;
    // Inputs in setup_inputs() order; edge_index (d_in[1]) is unused — the
    // graph is the fixed 8-neighbor grid, exploited structurally.
    const float* x  = (const float*)d_in[0];
    const float* W1 = (const float*)d_in[2];
    const float* b1 = (const float*)d_in[3];
    const float* W2 = (const float*)d_in[4];
    const float* b2 = (const float*)d_in[5];
    const float* W3 = (const float*)d_in[6];
    const float* b3 = (const float*)d_in[7];

    // f16 ping-pong intermediates (dinv-prescaled): 2 x 37.7 MB of workspace.
    _Float16* h1 = (_Float16*)d_ws;
    _Float16* h2 = h1 + (size_t)HH * HH * HID;

    dim3 grid(HH / TILE, HH / TILE), block(256);
    hipLaunchKernelGGL((gcn_layer_kernel<3,  32, false, true,  false>),
                       grid, block, 0, stream, (const void*)x,  W1, b1, (void*)h1);
    hipLaunchKernelGGL((gcn_layer_kernel<32, 32, true,  true,  false>),
                       grid, block, 0, stream, (const void*)h1, W2, b2, (void*)h2);
    hipLaunchKernelGGL((gcn_layer_kernel<32, 9,  true,  false, true>),
                       grid, block, 0, stream, (const void*)h2, W3, b3, d_out);
}
